// SelectiveScan2D_49022756716946
// MI455X (gfx1250) — matrix operand order repaired and checked by reading, not compiled, not run on hardware
//
#include <hip/hip_runtime.h>

// ---------------------------------------------------------------------------
// SelectiveScan2D for MI455X (gfx1250, wave32, WMMA).
// Memory-bound (~2.3 GB moved vs 48 GFLOP of GEMM); all channel-GEMMs run on
// v_wmma_f32_16x16x32_bf16 with fp32 accumulation. Scan operands are written
// once (alpha, u = sigmoid(beta)*z, gamma) and consumed by a wave-parallel
// shuffle scan (W axis) and a w-vectorized sequential scan (H axis).
// Round-3 tweaks: tile-uniform (scalar-branch) gate routing in the K1
// epilogue via readfirstlane, and one ds_load_b128 for the z values in the
// u = sigmoid(beta)*z path (was 8x ds_load_u16 under divergent exec masks).
// ---------------------------------------------------------------------------

#define DEVI static __device__ __forceinline__

typedef __attribute__((ext_vector_type(16))) __bf16 v16bf;
typedef __attribute__((ext_vector_type(8)))  float  v8f;

union FragAB { v16bf v; uint4 q[2]; };
union ZPack  { uint4 q; unsigned short s[8]; };

constexpr int Bn = 8, Cn = 192, Hn = 128, Wn = 128;
constexpr int HW = Hn * Wn;                       // 16384
// ---- workspace layout (bytes). Requires ws_size >= ~454 MB. ----
constexpr size_t WS_WPAR  = 0;                    // 576*192 bf16  = 221184
constexpr size_t WS_WLOC  = 221184;               // 192*192 bf16  =  73728
constexpr size_t WS_WOUT  = 294912;               // 192*192 bf16  =  73728
constexpr size_t WS_Z     = 368640;               // N*C bf16      = 50331648
constexpr size_t WS_ALPHA = 50700288;             // N*C f32       = 100663296
constexpr size_t WS_U     = 151363584;            // N*C f32
constexpr size_t WS_GAMMA = 252026880;            // N*C f32
constexpr size_t WS_YACC  = 352690176;            // N*C f32 -> end 453353472

DEVI float bf2f(unsigned short h) {
  __bf16 b; __builtin_memcpy(&b, &h, 2);
  return (float)b;
}
DEVI unsigned short f2bf(float f) {
  __bf16 b = (__bf16)f;                           // native cvt, RNE
  unsigned short u; __builtin_memcpy(&u, &b, 2);
  return u;
}
// Fast sigmoid: v_exp_f32 + v_rcp_f32 (1 ulp), no IEEE-divide expansion.
DEVI float sigmoidf(float x) { return __builtin_amdgcn_rcpf(1.0f + __expf(-x)); }
DEVI float siluf(float x)    { return x * sigmoidf(x); }

// ---------------------------------------------------------------------------
// K0: convert GEMM weights fp32 -> bf16 once (they then live in L2: 370 KB).
// ---------------------------------------------------------------------------
__global__ void wconvert_kernel(const float* __restrict__ par,
                                const float* __restrict__ loc,
                                const float* __restrict__ outw,
                                unsigned short* __restrict__ wpar,
                                unsigned short* __restrict__ wloc,
                                unsigned short* __restrict__ wout) {
  int i = blockIdx.x * 256 + threadIdx.x;
  if (i < 3 * Cn * Cn) wpar[i] = f2bf(par[i]);
  if (i < Cn * Cn) { wloc[i] = f2bf(loc[i]); wout[i] = f2bf(outw[i]); }
}

// ---------------------------------------------------------------------------
// K1: LayerNorm over C (per position) + gates GEMM (576x192 @ 192x32, WMMA).
// Block = 32 contiguous W positions at fixed (b,h); 256 threads = 8 waves.
// Stores z (bf16), alpha, u = sigmoid(beta)*z, gamma.
// ---------------------------------------------------------------------------
__global__ __launch_bounds__(256)
void ln_gates_kernel(const float* __restrict__ x,
                     const float* __restrict__ ln_w,
                     const float* __restrict__ ln_b,
                     const unsigned short* __restrict__ wpar,
                     const float* __restrict__ par_b,
                     unsigned short* __restrict__ zg,
                     float* __restrict__ alpha_g,
                     float* __restrict__ u_g,
                     float* __restrict__ gamma_g) {
  __shared__ __align__(16) float xs[192][33];           // staged x tile
  __shared__ __align__(16) unsigned short zls[32][200]; // z[pos][chan] bf16
  __shared__ float rs[8][32], rq[8][32];
  __shared__ float mean_s[32], rstd_s[32];
  __shared__ float pb_s[576];                           // par_b in LDS

  const int tid = threadIdx.x, lane = tid & 31, wid = tid >> 5;
  const int w0 = blockIdx.x * 32, h = blockIdx.y, b = blockIdx.z;
  const long long pbase = ((long long)b * Cn) * HW + (long long)h * Wn + w0;

  for (int i = tid; i < 576; i += 256) pb_s[i] = par_b[i];

  // Stage x, partial channel moments (coalesced: lanes sweep w).
  {
    const int p = tid & 31, chunk = tid >> 5;           // 8 chunks x 24 chans
    float s = 0.f, q = 0.f;
    #pragma unroll
    for (int i = 0; i < 24; ++i) {
      int c = chunk * 24 + i;
      float v = x[pbase + (long long)c * HW + p];
      xs[c][p] = v; s += v; q += v * v;
    }
    rs[chunk][p] = s; rq[chunk][p] = q;
  }
  __syncthreads();
  if (tid < 32) {
    float s = 0.f, q = 0.f;
    #pragma unroll
    for (int k = 0; k < 8; ++k) { s += rs[k][tid]; q += rq[k][tid]; }
    float m = s * (1.0f / 192.0f);
    float var = q * (1.0f / 192.0f) - m * m;
    mean_s[tid] = m; rstd_s[tid] = __frsqrt_rn(var + 1e-6f);
  }
  __syncthreads();
  for (int idx = tid; idx < 192 * 32; idx += 256) {
    int c = idx >> 5, p = idx & 31;
    float zv = (xs[c][p] - mean_s[p]) * rstd_s[p] * ln_w[c] + ln_b[c];
    unsigned short hb = f2bf(zv);
    zls[p][c] = hb;
    zg[pbase + (long long)c * HW + p] = hb;
  }
  __syncthreads();

  // gates = par_w @ z : 36 out-tiles x 2 pos-tiles = 72 WMMA tiles / 8 waves.
  const int kA = (lane < 16) ? 0 : 8;     // A-frag K chunk offset (ISA layout)
  const int kB = (lane < 16) ? 0 : 8;     // B-frag K chunk offset (ISA layout)
  const int ml = lane & 15;
  const int hi = (lane < 16) ? 0 : 8;     // C/D frag row offset
  for (int t = wid * 9; t < wid * 9 + 9; ++t) {
    int ot = t >> 1, pt = t & 1;
    v8f acc = {0, 0, 0, 0, 0, 0, 0, 0};
    const unsigned short* wrow = wpar + (ot * 16 + ml) * 192;
    const unsigned short* brow = &zls[pt * 16 + ml][0];
    __builtin_prefetch(wrow, 0, 0);                  // global_prefetch_b8
    #pragma unroll
    for (int k = 0; k < 192; k += 32) {
      FragAB a, bb;
      a.q[0]  = *(const uint4*)(wrow + k + kA);      // K = k .. k+7   (+8 hi)
      a.q[1]  = *(const uint4*)(wrow + k + 16 + kA); // K = k+16..k+23 (+8 hi)
      bb.q[0] = *(const uint4*)(brow + k + kB);      // K = k .. k+7   (+8 hi)
      bb.q[1] = *(const uint4*)(brow + k + 16 + kB); // K = k+16..k+23 (+8 hi)
      acc = __builtin_amdgcn_wmma_f32_16x16x32_bf16(
          false, a.v, false, bb.v, (short)0, acc, false, false);
    }
    const int pos = pt * 16 + ml;
    // Tile-uniform routing: a 16-row tile never straddles a 192-boundary.
    const int ots = __builtin_amdgcn_readfirstlane(ot);
    if (ots < 12) {                                   // ---- alpha tiles ----
      #pragma unroll
      for (int r = 0; r < 8; ++r) {
        int ch = ot * 16 + hi + r;
        alpha_g[pbase + (long long)ch * HW + pos] = sigmoidf(acc[r] + pb_s[ch]);
      }
    } else if (ots < 24) {                            // ---- u = sig(b)*z ----
      const int chb = (ot - 12) * 16 + hi;            // 16B-aligned in zls row
      ZPack zp; zp.q = *(const uint4*)(&zls[pos][chb]);
      #pragma unroll
      for (int r = 0; r < 8; ++r) {
        int ch = chb + r;
        float g = sigmoidf(acc[r] + pb_s[ch + 192]);
        u_g[pbase + (long long)ch * HW + pos] = g * bf2f(zp.s[r]);
      }
    } else {                                          // ---- gamma tiles ----
      #pragma unroll
      for (int r = 0; r < 8; ++r) {
        int ch = (ot - 24) * 16 + hi + r;
        gamma_g[pbase + (long long)ch * HW + pos] =
            sigmoidf(acc[r] + pb_s[ch + 384]);
      }
    }
  }
}

// ---------------------------------------------------------------------------
// K2: bidirectional affine scan along W (length 128) — one wave per row.
// Lane owns 4 contiguous elements; wave-level (A,S) Hillis-Steele scan via
// shuffles (wave32). Writes y_acc = 0.5*gamma*(s_fwd + s_bwd).
// ---------------------------------------------------------------------------
__global__ __launch_bounds__(256)
void scan_w_kernel(const float* __restrict__ alpha_g,
                   const float* __restrict__ u_g,
                   const float* __restrict__ gamma_g,
                   float* __restrict__ y_acc) {
  const int lane = threadIdx.x & 31, wid = threadIdx.x >> 5;
  const long long row = (long long)blockIdx.x * 8 + wid;   // B*C*H rows
  const long long base = row * Wn + lane * 4;
  const float4 a4 = *(const float4*)(alpha_g + base);
  const float4 u4 = *(const float4*)(u_g + base);
  const float4 g4 = *(const float4*)(gamma_g + base);
  float av[4] = {a4.x, a4.y, a4.z, a4.w};
  float uv[4] = {u4.x, u4.y, u4.z, u4.w};
  float gv[4] = {g4.x, g4.y, g4.z, g4.w};

  // ---- forward ----
  float A = 1.f, S = 0.f;
  #pragma unroll
  for (int i = 0; i < 4; ++i) { S = av[i] * S + uv[i]; A *= av[i]; }
  #pragma unroll
  for (int d = 1; d < 32; d <<= 1) {
    float So = __shfl_up(S, d, 32), Ao = __shfl_up(A, d, 32);
    if (lane >= d) { S = A * So + S; A = A * Ao; }
  }
  float carry = __shfl_up(S, 1, 32);
  if (lane == 0) carry = 0.f;
  float sf[4], s = carry;
  #pragma unroll
  for (int i = 0; i < 4; ++i) { s = av[i] * s + uv[i]; sf[i] = s; }

  // ---- backward ----
  A = 1.f; S = 0.f;
  #pragma unroll
  for (int i = 3; i >= 0; --i) { S = av[i] * S + uv[i]; A *= av[i]; }
  #pragma unroll
  for (int d = 1; d < 32; d <<= 1) {
    float So = __shfl_down(S, d, 32), Ao = __shfl_down(A, d, 32);
    if (lane + d < 32) { S = A * So + S; A = A * Ao; }
  }
  carry = __shfl_down(S, 1, 32);
  if (lane == 31) carry = 0.f;
  s = carry;
  float sb[4];
  #pragma unroll
  for (int i = 3; i >= 0; --i) { s = av[i] * s + uv[i]; sb[i] = s; }

  float4 o;
  o.x = 0.5f * gv[0] * (sf[0] + sb[0]);
  o.y = 0.5f * gv[1] * (sf[1] + sb[1]);
  o.z = 0.5f * gv[2] * (sf[2] + sb[2]);
  o.w = 0.5f * gv[3] * (sf[3] + sb[3]);
  *(float4*)(y_acc + base) = o;
}

// ---------------------------------------------------------------------------
// K3: bidirectional scan along H. 128 threads sweep all w at fixed (b,c);
// fully coalesced 512B rows; y_acc += 0.5*gamma*s (fwd then bwd).
// ---------------------------------------------------------------------------
__global__ __launch_bounds__(128)
void scan_h_kernel(const float* __restrict__ alpha_g,
                   const float* __restrict__ u_g,
                   const float* __restrict__ gamma_g,
                   float* __restrict__ y_acc) {
  const long long base = (long long)blockIdx.x * HW + threadIdx.x;
  float s = 0.f;
  for (int h = 0; h < Hn; ++h) {
    long long i = base + (long long)h * Wn;
    s = alpha_g[i] * s + u_g[i];
    y_acc[i] += 0.5f * gamma_g[i] * s;
  }
  s = 0.f;
  for (int h = Hn - 1; h >= 0; --h) {
    long long i = base + (long long)h * Wn;
    s = alpha_g[i] * s + u_g[i];
    y_acc[i] += 0.5f * gamma_g[i] * s;
  }
}

// ---------------------------------------------------------------------------
// K4: depthwise 3x3 on z (halo-staged in LDS) -> loc GEMM (+y_acc, SiLU)
//     -> out GEMM (+bias, +x residual) -> final output. Both GEMMs on WMMA.
// ---------------------------------------------------------------------------
__global__ __launch_bounds__(256)
void local_out_kernel(const float* __restrict__ x,
                      const unsigned short* __restrict__ zg,
                      const float* __restrict__ dw_w,
                      const unsigned short* __restrict__ wloc,
                      const float* __restrict__ loc_b,
                      const unsigned short* __restrict__ wout,
                      const float* __restrict__ out_b,
                      const float* __restrict__ y_acc,
                      float* __restrict__ out) {
  __shared__ unsigned short zt[192][3][34];             // z halo (bf16)
  __shared__ __align__(16) unsigned short dls[32][200]; // dwconv result
  __shared__ __align__(16) unsigned short yls[32][200]; // silu(local+scan)
  __shared__ float lb_s[192], ob_s[192];                // biases in LDS
  const int tid = threadIdx.x, lane = tid & 31, wid = tid >> 5;
  const int w0 = blockIdx.x * 32, h = blockIdx.y, b = blockIdx.z;
  const long long pbase = ((long long)b * Cn) * HW + (long long)h * Wn + w0;

  if (tid < 192) { lb_s[tid] = loc_b[tid]; ob_s[tid] = out_b[tid]; }

  // Stage z halo rows h-1..h+1, w0-1..w0+32 (zero-padded at borders).
  for (int idx = tid; idx < 192 * 3 * 34; idx += 256) {
    int c = idx / 102, rem = idx - c * 102;
    int dy = rem / 34, j = rem - dy * 34;
    int hh = h - 1 + dy, ww = w0 - 1 + j;
    unsigned short v = 0;
    if ((unsigned)hh < (unsigned)Hn && (unsigned)ww < (unsigned)Wn)
      v = zg[((long long)b * Cn + c) * HW + (long long)hh * Wn + ww];
    zt[c][dy][j] = v;
  }
  __syncthreads();

  // Depthwise 3x3 (fp32 math, bf16 result into B-operand layout [pos][chan]).
  for (int t = tid; t < 192 * 32; t += 256) {
    int c = t >> 5, p = t & 31;
    const float* kw = dw_w + c * 9;
    float acc = 0.f;
    #pragma unroll
    for (int dy = 0; dy < 3; ++dy)
      #pragma unroll
      for (int dx = 0; dx < 3; ++dx)
        acc += kw[dy * 3 + dx] * bf2f(zt[c][dy][p + dx]);
    dls[p][c] = f2bf(acc);
  }
  __syncthreads();

  const int kA = (lane < 16) ? 0 : 8;
  const int kB = (lane < 16) ? 0 : 8;
  const int ml = lane & 15;
  const int hi = (lane < 16) ? 0 : 8;

  // GEMM1: local = loc_w @ d ; fuse +loc_b +y_acc and SiLU -> yls (bf16).
  for (int t = wid * 3; t < wid * 3 + 3; ++t) {
    int ot = t >> 1, pt = t & 1;                 // 12 x 2 = 24 tiles
    v8f acc = {0, 0, 0, 0, 0, 0, 0, 0};
    const unsigned short* wrow = wloc + (ot * 16 + ml) * 192;
    const unsigned short* brow = &dls[pt * 16 + ml][0];
    #pragma unroll
    for (int k = 0; k < 192; k += 32) {
      FragAB a, bb;
      a.q[0]  = *(const uint4*)(wrow + k + kA);
      a.q[1]  = *(const uint4*)(wrow + k + 16 + kA);
      bb.q[0] = *(const uint4*)(brow + k + kB);
      bb.q[1] = *(const uint4*)(brow + k + 16 + kB);
      acc = __builtin_amdgcn_wmma_f32_16x16x32_bf16(
          false, a.v, false, bb.v, (short)0, acc, false, false);
    }
    int pos = pt * 16 + ml;
    #pragma unroll
    for (int r = 0; r < 8; ++r) {
      int och = ot * 16 + hi + r;
      long long gi = pbase + (long long)och * HW + pos;
      float y = acc[r] + lb_s[och] + y_acc[gi];
      yls[pos][och] = f2bf(siluf(y));
    }
  }
  __syncthreads();

  // GEMM2: out = out_w @ silu(y) ; + out_b + x residual -> d_out.
  for (int t = wid * 3; t < wid * 3 + 3; ++t) {
    int ot = t >> 1, pt = t & 1;
    v8f acc = {0, 0, 0, 0, 0, 0, 0, 0};
    const unsigned short* wrow = wout + (ot * 16 + ml) * 192;
    const unsigned short* brow = &yls[pt * 16 + ml][0];
    #pragma unroll
    for (int k = 0; k < 192; k += 32) {
      FragAB a, bb;
      a.q[0]  = *(const uint4*)(wrow + k + kA);
      a.q[1]  = *(const uint4*)(wrow + k + 16 + kA);
      bb.q[0] = *(const uint4*)(brow + k + kB);
      bb.q[1] = *(const uint4*)(brow + k + 16 + kB);
      acc = __builtin_amdgcn_wmma_f32_16x16x32_bf16(
          false, a.v, false, bb.v, (short)0, acc, false, false);
    }
    int pos = pt * 16 + ml;
    #pragma unroll
    for (int r = 0; r < 8; ++r) {
      int och = ot * 16 + hi + r;
      long long gi = pbase + (long long)och * HW + pos;
      out[gi] = acc[r] + ob_s[och] + x[gi];
    }
  }
}

// ---------------------------------------------------------------------------
extern "C" void kernel_launch(void* const* d_in, const int* in_sizes, int n_in,
                              void* d_out, int out_size, void* d_ws,
                              size_t ws_size, hipStream_t stream) {
  const float* x     = (const float*)d_in[0];
  const float* ln_w  = (const float*)d_in[1];
  const float* ln_b  = (const float*)d_in[2];
  const float* dw_w  = (const float*)d_in[3];
  const float* loc_w = (const float*)d_in[4];
  const float* loc_b = (const float*)d_in[5];
  const float* par_w = (const float*)d_in[6];
  const float* par_b = (const float*)d_in[7];
  const float* out_w = (const float*)d_in[8];
  const float* out_b = (const float*)d_in[9];
  float* out = (float*)d_out;

  char* ws = (char*)d_ws;
  unsigned short* wpar  = (unsigned short*)(ws + WS_WPAR);
  unsigned short* wloc  = (unsigned short*)(ws + WS_WLOC);
  unsigned short* wout  = (unsigned short*)(ws + WS_WOUT);
  unsigned short* zg    = (unsigned short*)(ws + WS_Z);
  float* alpha = (float*)(ws + WS_ALPHA);
  float* u     = (float*)(ws + WS_U);
  float* gamma = (float*)(ws + WS_GAMMA);
  float* yacc  = (float*)(ws + WS_YACC);

  (void)in_sizes; (void)n_in; (void)out_size; (void)ws_size;

  // K0: weight conversion (covers 3*C*C = 110592 par elems and C*C others).
  wconvert_kernel<<<(3 * Cn * Cn + 255) / 256, 256, 0, stream>>>(
      par_w, loc_w, out_w, wpar, wloc, wout);

  dim3 gtile(Wn / 32, Hn, Bn);  // 4 x 128 x 8 blocks, 32 positions each
  ln_gates_kernel<<<gtile, 256, 0, stream>>>(
      x, ln_w, ln_b, wpar, par_b, zg, alpha, u, gamma);

  scan_w_kernel<<<(Bn * Cn * Hn) / 8, 256, 0, stream>>>(alpha, u, gamma, yacc);
  scan_h_kernel<<<Bn * Cn, 128, 0, stream>>>(alpha, u, gamma, yacc);

  local_out_kernel<<<gtile, 256, 0, stream>>>(
      x, zg, dw_w, wloc, loc_b, wout, out_b, yacc, out);
}
